// PassModel_MSAGEV_52785148068159
// MI455X (gfx1250) — matrix operand order repaired and checked by reading, not compiled, not run on hardware
//
#include <hip/hip_runtime.h>
#include <hip/hip_bf16.h>

// ---------------------------------------------------------------------------
// MI455X (gfx1250, wave32) implementation of PassModel_MSAGEV forward.
// Heavy work = ConvGRU convs -> bf16 WMMA implicit GEMM (v_wmma_f32_16x16x32_bf16),
// NHWC bf16 activations padded to 64ch, weights pre-packed into B-fragment
// layout and staged through LDS via GLOBAL_LOAD_ASYNC_TO_LDS (ASYNCcnt path).
// Conv kernels templated on (KH,KW,MODE) for fully-unrolled WMMA tap chains.
// ---------------------------------------------------------------------------

typedef __bf16 bf16;
typedef __attribute__((ext_vector_type(16))) __bf16 v16bf;
typedef __attribute__((ext_vector_type(8)))  float  v8f;
typedef int vint4 __attribute__((vector_size(16)));   // raw clang vector (not HIP_vector_type)

static constexpr int IMGH = 192, IMGW = 192;
static constexpr int HW   = IMGH * IMGW;      // 36864 pixels per cam
static constexpr int NCAM = 4;
static constexpr long NPIX = (long)NCAM * HW; // 147456
static constexpr int CPAD = 64;               // padded input channels for conv
static constexpr int NNODE = 23;
static constexpr int NQ = 4096;

__device__ __forceinline__ float sigmf(float x) { return 1.f / (1.f + __expf(-x)); }

// ---------------------------------------------------------------------------
// Cooperative global->LDS staging. Prefers CDNA5 async-to-LDS (ASYNCcnt),
// falls back to VGPR round-trip if the builtin is unavailable.
// bytes must be a multiple of 16. Caller must __syncthreads() afterwards.
// ---------------------------------------------------------------------------
__device__ __forceinline__ void stage_to_lds(const void* gsrc, void* lds_dst, int bytes) {
  const int n16 = bytes >> 4;
#if __has_builtin(__builtin_amdgcn_global_load_async_to_lds_b128)
  typedef __attribute__((address_space(1))) vint4 gv4;
  typedef __attribute__((address_space(3))) vint4 lv4;
  gv4* g = (gv4*)gsrc;
  lv4* l = (lv4*)lds_dst;
  for (int i = threadIdx.x; i < n16; i += blockDim.x)
    __builtin_amdgcn_global_load_async_to_lds_b128(g + i, l + i, 0, 0);
#if __has_builtin(__builtin_amdgcn_s_wait_asynccnt)
  __builtin_amdgcn_s_wait_asynccnt(0);
#else
  asm volatile("s_wait_asynccnt 0" ::: "memory");
#endif
#else
  const int4* g = (const int4*)gsrc;
  int4* l = (int4*)lds_dst;
  for (int i = threadIdx.x; i < n16; i += blockDim.x) l[i] = g[i];
#endif
}

// ---------------------------------------------------------------------------
// zero fill
// ---------------------------------------------------------------------------
__global__ void k_zero_bf16(bf16* p, long n) {
  long i  = (long)blockIdx.x * blockDim.x + threadIdx.x;
  long st = (long)gridDim.x * blockDim.x;
  for (; i < n; i += st) p[i] = (bf16)0.f;
}

// ---------------------------------------------------------------------------
// Pack f32 weights W[cout][cin][kh][kw] into WMMA B-fragment layout:
//   dst[nt][tap][kc][lane(32)][half(16)]  (bf16)
// lane<16 holds K = kc*32 + 0..15 for column n = nt*16 + lane,
// lane>=16 holds K = kc*32 + 16..31 for column n = nt*16 + lane-16.
// Zero-pads K to CP and cout to NT*16.
// ---------------------------------------------------------------------------
__global__ void k_pack_B(const float* __restrict__ W, bf16* __restrict__ dst,
                         int Cout, int Cin, int KH, int KW, int CP) {
  const int taps = KH * KW, KC = CP / 32, NT = (Cout + 15) / 16;
  const long total = (long)NT * taps * KC * 512;
  long id = (long)blockIdx.x * blockDim.x + threadIdx.x;
  long st = (long)gridDim.x * blockDim.x;
  for (; id < total; id += st) {
    int half = (int)(id & 15);
    int lane = (int)((id >> 4) & 31);
    long rest = id >> 9;
    int kc  = (int)(rest % KC);  rest /= KC;
    int tap = (int)(rest % taps);
    int nt  = (int)(rest / taps);
    int n = nt * 16 + (lane & 15);
    int kk = ((lane >> 4) << 4) + half;   // 0..31 within chunk
    int k = kc * 32 + kk;                 // ci
    float v = 0.f;
    if (n < Cout && k < Cin) v = W[((long)n * Cin + k) * taps + tap];
    dst[id] = (bf16)v;
  }
}

// ---------------------------------------------------------------------------
// Build concat input buffers (bf16 NHWC, CPAD channels).
// layer0: x = current frame pixels (3 ch, from f32 img), h = H1 (32 ch)
// ---------------------------------------------------------------------------
__global__ void k_build_l0(const float* __restrict__ img, int t,
                           const bf16* __restrict__ H1,
                           bf16* __restrict__ INA, bf16* __restrict__ INB) {
  long id = (long)blockIdx.x * blockDim.x + threadIdx.x;  // cam*HW + pix
  if (id >= NPIX) return;
  int cam = (int)(id / HW), p = (int)(id % HW);
  const float* px = img + (((long)cam * 8 + t) * HW + p) * 3;
  long o = id * CPAD;
  for (int c = 0; c < 3; ++c) { bf16 v = (bf16)px[c]; INA[o + c] = v; INB[o + c] = v; }
  const bf16* h = H1 + id * 32;
  for (int c = 0; c < 32; ++c) INA[o + 3 + c] = h[c];
  for (int c = 35; c < CPAD; ++c) { INA[o + c] = (bf16)0.f; INB[o + c] = (bf16)0.f; }
}

// layers 1/2: x = previous layer hidden (xch), h = this layer hidden (hch)
__global__ void k_build_hh(const bf16* __restrict__ X, int xch,
                           const bf16* __restrict__ Hs, int hch,
                           bf16* __restrict__ INA, bf16* __restrict__ INB) {
  long id = (long)blockIdx.x * blockDim.x + threadIdx.x;
  if (id >= NPIX) return;
  long o = id * CPAD;
  for (int c = 0; c < xch; ++c) { bf16 v = X[id * xch + c]; INA[o + c] = v; INB[o + c] = v; }
  for (int c = 0; c < hch; ++c) INA[o + xch + c] = Hs[id * hch + c];
  for (int c = xch + hch; c < CPAD; ++c) { INA[o + c] = (bf16)0.f; INB[o + c] = (bf16)0.f; }
}

// ---------------------------------------------------------------------------
// WMMA implicit-GEMM convolution, SAME padding, stride 1, NHWC bf16 (CPAD=64).
// Each wave: 16 consecutive pixels in one row x 16 output channels.
// Block = 4 waves (4 pixel strips), one cout-tile; B fragments staged in LDS
// via async-to-LDS. KH,KW,MODE compile-time -> fully unrolled wmma chain
// (18 wmma for 3x3, 50 for 5x5).
// MODE 0 (gates): g = sigmoid(conv+bg); co<hd -> write r*h into INB h-region;
//                 co>=hd -> write z into Z.
// MODE 1 (cand):  n = tanh(conv+bc); h = (1-z)*h + z*n  (in-place H update).
// ---------------------------------------------------------------------------
template <int KH, int KW, int MODE>
__global__ void k_conv_wmma(const bf16* __restrict__ IN, const bf16* __restrict__ Wp,
                            const float* __restrict__ bias,
                            int hd, int xch,
                            bf16* __restrict__ Z, bf16* __restrict__ INB,
                            bf16* __restrict__ H) {
  extern __shared__ int4 sB[];
  constexpr int taps = KH * KW, KC = 2;
  constexpr int PH = KH / 2, PW = KW / 2;
  const int nt = blockIdx.y, cam = blockIdx.z;
  stage_to_lds((const int4*)Wp + (long)nt * taps * KC * 64, sB, taps * KC * 1024);
  __syncthreads();

  const int lane = threadIdx.x & 31;
  const int wave = threadIdx.x >> 5;
  const int strip = blockIdx.x * 4 + wave;        // 2304 strips per cam
  const int y = strip / 12, x0 = (strip % 12) * 16;
  const int ml = lane & 15;                       // A-row (M) for this lane
  const int ko = (lane < 16) ? 0 : 16;            // A K-offset per half-wave
  const bf16* camIN = IN + (long)cam * HW * CPAD;

  v8f acc = {0.f, 0.f, 0.f, 0.f, 0.f, 0.f, 0.f, 0.f};
#pragma unroll
  for (int dy = 0; dy < KH; ++dy) {
    const int yin = y + dy - PH;
    const bool rowOK = (unsigned)yin < (unsigned)IMGH;
    const bf16* rowPtr = camIN + (long)yin * IMGW * CPAD;
#pragma unroll
    for (int dx = 0; dx < KW; ++dx) {
      const int xin = x0 + ml + dx - PW;
      const bool valid = rowOK && ((unsigned)xin < (unsigned)IMGW);
      const bf16* ap0 = rowPtr + (long)xin * CPAD;
      const int tp = dy * KW + dx;
#pragma unroll
      for (int kc = 0; kc < KC; ++kc) {
        union { int4 q[2]; v16bf v; } A;
        if (valid) {
          A.q[0] = *(const int4*)(ap0 + kc * 32 + ko);
          A.q[1] = *(const int4*)(ap0 + kc * 32 + ko + 8);
        } else {
          A.q[0] = make_int4(0, 0, 0, 0);
          A.q[1] = make_int4(0, 0, 0, 0);
        }
        union { int4 q[2]; v16bf v; } B;
        const int4* sp = sB + (long)(tp * KC + kc) * 64 + lane * 2;
        B.q[0] = sp[0]; B.q[1] = sp[1];
        acc = __builtin_amdgcn_wmma_f32_16x16x32_bf16(false, A.v, false, B.v,
                                                      (short)0, acc, false, false);
      }
    }
  }

  // epilogue (C/D layout: lane<16 -> M=j, lane>=16 -> M=8+j; N = lane&15)
  const int n = lane & 15;
  const int co = nt * 16 + n;
  const int mbase = (lane >> 4) * 8;
  const long pixRow = (long)cam * HW + (long)y * IMGW + x0;
  const float b = bias[co];
  if (MODE == 0) {
    const bool isR = co < hd;
#pragma unroll
    for (int j = 0; j < 8; ++j) {
      long pix = pixRow + mbase + j;
      float sg = sigmf(acc[j] + b);
      if (isR) {
        float hold = (float)H[pix * hd + co];
        INB[pix * CPAD + xch + co] = (bf16)(sg * hold);   // r * h
      } else {
        Z[pix * hd + (co - hd)] = (bf16)sg;               // z
      }
    }
  } else {
#pragma unroll
    for (int j = 0; j < 8; ++j) {
      long pix = pixRow + mbase + j;
      float nv = tanhf(acc[j] + b);
      float z = (float)Z[pix * hd + co];
      float hold = (float)H[pix * hd + co];
      H[pix * hd + co] = (bf16)((1.f - z) * hold + z * nv);
    }
  }
}

// ---------------------------------------------------------------------------
// Single-block GRU layer scan (T=200, 23 nodes, H=16). torch gate order r,z,n.
// mode 0: input = x [23][200][3]; mode 1: input = seq [t][23][16];
// mode 2: input = concat(ball, x) (6 feats, ball = node0).
// ---------------------------------------------------------------------------
__global__ void k_gru(const float* __restrict__ xin, int mode, int IN,
                      const float* __restrict__ Wih, const float* __restrict__ Whh,
                      const float* __restrict__ bih, const float* __restrict__ bhh,
                      float* __restrict__ seqout, float* __restrict__ finout) {
  const int T = 200;
  __shared__ float h[NNODE * 16];
  const int tid = threadIdx.x;
  const int node = tid >> 4, hu = tid & 15;
  const bool act = tid < NNODE * 16;
  if (act) h[tid] = 0.f;
  __syncthreads();
  for (int t = 0; t < T; ++t) {
    float hnew = 0.f;
    if (act) {
      float xv[16];
      for (int k = 0; k < IN; ++k) {
        float v;
        if (mode == 0)      v = xin[node * 600 + t * 3 + k];
        else if (mode == 1) v = xin[((long)t * NNODE + node) * 16 + k];
        else                v = (k < 3) ? xin[t * 3 + k]
                                        : xin[node * 600 + t * 3 + (k - 3)];
        xv[k] = v;
      }
      float ir = bih[hu], iz = bih[16 + hu], inn = bih[32 + hu];
      for (int k = 0; k < IN; ++k) {
        ir  += Wih[hu * IN + k] * xv[k];
        iz  += Wih[(16 + hu) * IN + k] * xv[k];
        inn += Wih[(32 + hu) * IN + k] * xv[k];
      }
      float hr = bhh[hu], hz = bhh[16 + hu], hn = bhh[32 + hu];
      for (int k = 0; k < 16; ++k) {
        float hv = h[node * 16 + k];
        hr += Whh[hu * 16 + k] * hv;
        hz += Whh[(16 + hu) * 16 + k] * hv;
        hn += Whh[(32 + hu) * 16 + k] * hv;
      }
      float r = sigmf(ir + hr), z = sigmf(iz + hz);
      float nn = tanhf(inn + r * hn);
      hnew = (1.f - z) * nn + z * h[tid];
    }
    __syncthreads();
    if (act) {
      h[tid] = hnew;
      if (seqout) seqout[((long)t * NNODE + node) * 16 + hu] = hnew;
    }
    __syncthreads();
  }
  if (act) finout[node * 16 + hu] = h[tid];
}

// ---------------------------------------------------------------------------
// AdaptiveMaxPool2d(3) over H3 [4][192][192][16] bf16 -> video_feat [4][144]
// feature index = c*9 + py*3 + px (channel-major reshape).
// ---------------------------------------------------------------------------
__global__ void k_pool(const bf16* __restrict__ H3, float* __restrict__ vfeat) {
  const int b = blockIdx.x;           // 36 = 4 cams * 9 cells
  const int cam = b / 9, cell = b % 9, py = cell / 3, px = cell % 3;
  const int tid = threadIdx.x;
  const int ch = tid & 15, part = tid >> 4;   // 16 parts of 4 rows each
  float m = -1e30f;
  for (int ry = 0; ry < 4; ++ry) {
    int y = py * 64 + part * 4 + ry;
    for (int x = 0; x < 64; ++x) {
      int xx = px * 64 + x;
      float v = (float)H3[((long)cam * HW + (long)y * IMGW + xx) * 16 + ch];
      m = fmaxf(m, v);
    }
  }
  __shared__ float s[256];
  s[tid] = m;
  __syncthreads();
  if (part == 0) {
    float r = s[ch];
    for (int pp = 1; pp < 16; ++pp) r = fmaxf(r, s[pp * 16 + ch]);
    vfeat[cam * 144 + ch * 9 + py * 3 + px] = r;
  }
}

// node_in [23][192] = [traj_feat(48) | v_feat(144)]
__global__ void k_node_in(const float* __restrict__ naF, const float* __restrict__ vfeat,
                          const int* __restrict__ imginds, float* __restrict__ node_in) {
  int id = blockIdx.x * blockDim.x + threadIdx.x;
  if (id >= NNODE * 192) return;
  int n = id / 192, c = id % 192;
  float v;
  if (c < 48) { int l = c >> 4, hu = c & 15; v = naF[l * 368 + n * 16 + hu]; }
  else {
    int cv = c - 48; v = 0.f;
    for (int cam = 0; cam < 4; ++cam)
      if (imginds[cam] == n) v = vfeat[cam * 144 + cv];
  }
  node_in[id] = v;
}

// ---------------------------------------------------------------------------
// SAGE layer 1 (att=3), 4 heads. Full graph minus self-loops:
// edge (src=j -> dst=i) for all j != i.
// ---------------------------------------------------------------------------
__global__ void k_sage1(const float* __restrict__ node_in, const float* __restrict__ att,
                        const float* fc0, const float* aw0, const float* ab0,
                        const float* fc1, const float* aw1, const float* ab1,
                        const float* fc2, const float* aw2, const float* ab2,
                        const float* fc3, const float* aw3, const float* ab3,
                        float* __restrict__ h1) {
  const float* fc[4] = {fc0, fc1, fc2, fc3};
  const float* aw[4] = {aw0, aw1, aw2, aw3};
  const float* ab[4] = {ab0, ab1, ab2, ab3};
  const int i = blockIdx.x, tid = threadIdx.x;
  __shared__ float dd[NNODE * 4];
  __shared__ float ss[4];
  __shared__ float f[4 * 192];
  if (tid < NNODE * 4) {
    int j = tid >> 2, hh = tid & 3;
    float a = 0.f;
    for (int k = 0; k < 16; ++k) a += att[j * 16 + k] * aw[hh][k];
    dd[tid] = a;
  }
  if (tid >= 96 && tid < 100) {
    int hh = tid - 96;
    float a = 0.f;
    for (int k = 0; k < 16; ++k) a += att[i * 16 + k] * aw[hh][16 + k];
    ss[hh] = a;
  }
  __syncthreads();
  if (tid < 192) {
    for (int hh = 0; hh < 4; ++hh) {
      float a = 0.f;
      for (int j = 0; j < NNODE; ++j) {
        if (j == i) continue;
        float e = sigmf(dd[j * 4 + hh] + ss[hh] + ab[hh][0]);
        a += e * node_in[j * 192 + tid];
      }
      f[hh * 192 + tid] = a;
    }
  }
  __syncthreads();
  if (tid < 96) {
    int hh = tid / 24, co = tid % 24;
    const float* W = fc[hh];
    float a = 0.f;
    for (int c = 0; c < 192; ++c)
      a += f[hh * 192 + c] * W[co * 384 + c] + node_in[i * 192 + c] * W[co * 384 + 192 + c];
    h1[i * 96 + hh * 24 + co] = a;
  }
}

// ELU then SAGE layer 2 (att=0, no bias)
__global__ void k_sage2(const float* __restrict__ h1, const float* __restrict__ fcW2,
                        const float* __restrict__ attW2, float* __restrict__ g) {
  const int i = blockIdx.x, tid = threadIdx.x;
  __shared__ float hx[NNODE * 96];
  __shared__ float d2[NNODE];
  __shared__ float s2v;
  __shared__ float f2[96];
  for (int id = tid; id < NNODE * 96; id += blockDim.x) {
    float v = h1[id];
    hx[id] = v > 0.f ? v : (__expf(v) - 1.f);   // elu
  }
  __syncthreads();
  if (tid < NNODE) {
    float a = 0.f;
    for (int c = 0; c < 96; ++c) a += hx[tid * 96 + c] * attW2[c];
    d2[tid] = a;
  }
  if (tid == 32) {
    float a = 0.f;
    for (int c = 0; c < 96; ++c) a += hx[i * 96 + c] * attW2[96 + c];
    s2v = a;
  }
  __syncthreads();
  if (tid < 96) {
    float a = 0.f;
    for (int j = 0; j < NNODE; ++j) {
      if (j == i) continue;
      a += sigmf(d2[j] + s2v) * hx[j * 96 + tid];
    }
    f2[tid] = a;
  }
  __syncthreads();
  if (tid < 96) {
    float a = 0.f;
    for (int c = 0; c < 96; ++c)
      a += f2[c] * fcW2[tid * 192 + c] + hx[i * 96 + c] * fcW2[tid * 192 + 96 + c];
    g[i * 96 + tid] = a;
  }
}

// union [4096][224] bf16
__global__ void k_union(const float* __restrict__ g, const float* __restrict__ traj2,
                        const int* __restrict__ qf, const int* __restrict__ qt,
                        bf16* __restrict__ U) {
  long id = (long)blockIdx.x * blockDim.x + threadIdx.x;
  if (id >= (long)NQ * 224) return;
  int q = (int)(id / 224), c = (int)(id % 224);
  float v;
  if (c < 96)       v = g[qf[q] * 96 + c];
  else if (c < 192) v = g[qt[q] * 96 + (c - 96)];
  else if (c < 208) v = traj2[qf[q] * 16 + (c - 192)];
  else              v = traj2[qt[q] * 16 + (c - 208)];
  U[id] = (bf16)v;
}

// MLP layer 1 via WMMA: [4096,224] @ [224,256] + b1, relu -> bf16 [4096,256]
__global__ void k_mlp1(const bf16* __restrict__ U, const bf16* __restrict__ Wp,
                       const float* __restrict__ b1, bf16* __restrict__ act1) {
  __shared__ int4 sB[7 * 64];
  const int nt = blockIdx.y;
  stage_to_lds((const int4*)Wp + (long)nt * 7 * 64, sB, 7 * 1024);
  __syncthreads();
  const int lane = threadIdx.x & 31, wave = threadIdx.x >> 5;
  const int row0 = (blockIdx.x * 4 + wave) * 16;
  const int ml = lane & 15, ko = (lane < 16) ? 0 : 16;
  const bf16* ap0 = U + (long)(row0 + ml) * 224;
  v8f acc = {0.f, 0.f, 0.f, 0.f, 0.f, 0.f, 0.f, 0.f};
#pragma unroll
  for (int kc = 0; kc < 7; ++kc) {
    union { int4 q[2]; v16bf v; } A;
    A.q[0] = *(const int4*)(ap0 + kc * 32 + ko);
    A.q[1] = *(const int4*)(ap0 + kc * 32 + ko + 8);
    union { int4 q[2]; v16bf v; } B;
    const int4* sp = sB + kc * 64 + lane * 2;
    B.q[0] = sp[0]; B.q[1] = sp[1];
    acc = __builtin_amdgcn_wmma_f32_16x16x32_bf16(false, A.v, false, B.v,
                                                  (short)0, acc, false, false);
  }
  const int n = lane & 15, co = nt * 16 + n, mbase = (lane >> 4) * 8;
  const float b = b1[co];
#pragma unroll
  for (int j = 0; j < 8; ++j) {
    int row = row0 + mbase + j;
    float v = acc[j] + b;
    act1[(long)row * 256 + co] = (bf16)(v > 0.f ? v : 0.f);
  }
}

// MLP layer 2 + sigmoid -> f32 out [4096]
__global__ void k_mlp2(const bf16* __restrict__ act1, const float* __restrict__ W2,
                       const float* __restrict__ b2, float* __restrict__ out) {
  int q = blockIdx.x * blockDim.x + threadIdx.x;
  if (q >= NQ) return;
  float a = b2[0];
  for (int k = 0; k < 256; ++k) a += (float)act1[(long)q * 256 + k] * W2[k];
  out[q] = sigmf(a);
}

// ---------------------------------------------------------------------------
// Host orchestration
// ---------------------------------------------------------------------------
extern "C" void kernel_launch(void* const* d_in, const int* in_sizes, int n_in,
                              void* d_out, int out_size, void* d_ws, size_t ws_size,
                              hipStream_t stream) {
  (void)in_sizes; (void)n_in; (void)out_size; (void)ws_size;
  const float* x   = (const float*)d_in[0];
  const float* img = (const float*)d_in[1];
  const int* q_from  = (const int*)d_in[4];
  const int* q_to    = (const int*)d_in[5];
  const int* imginds = (const int*)d_in[6];
  auto F = [&](int i) { return (const float*)d_in[i]; };

  // workspace carve (bump allocator, 256B aligned); total ~76 MB
  char* w = (char*)d_ws;
  auto alloc = [&](size_t b) -> char* {
    char* p = w; w += (b + 255) & ~(size_t)255; return p;
  };
  bf16* H1  = (bf16*)alloc(NPIX * 32 * 2);
  bf16* H2  = (bf16*)alloc(NPIX * 32 * 2);
  bf16* H3  = (bf16*)alloc(NPIX * 16 * 2);
  bf16* INA = (bf16*)alloc(NPIX * CPAD * 2);
  bf16* INB = (bf16*)alloc(NPIX * CPAD * 2);
  bf16* Zb  = (bf16*)alloc(NPIX * 32 * 2);
  bf16* pg0 = (bf16*)alloc(73728);
  bf16* pc0 = (bf16*)alloc(36864);
  bf16* pg1 = (bf16*)alloc(204800);
  bf16* pc1 = (bf16*)alloc(102400);
  bf16* pg2 = (bf16*)alloc(36864);
  bf16* pc2 = (bf16*)alloc(18432);
  bf16* pW1 = (bf16*)alloc(114688);
  float* seqA = (float*)alloc(200L * 23 * 16 * 4);
  float* seqB = (float*)alloc(200L * 23 * 16 * 4);
  float* naF  = (float*)alloc(3 * 368 * 4);
  float* taF  = (float*)alloc(368 * 4);
  float* aaF  = (float*)alloc(368 * 4);
  float* vfeat   = (float*)alloc(576 * 4);
  float* node_in = (float*)alloc(23 * 192 * 4);
  float* h1  = (float*)alloc(23 * 96 * 4);
  float* gf  = (float*)alloc(23 * 96 * 4);
  bf16* U  = (bf16*)alloc((size_t)NQ * 224 * 2);
  bf16* A1 = (bf16*)alloc((size_t)NQ * 256 * 2);

  // zero hidden states
  k_zero_bf16<<<2048, 256, 0, stream>>>(H1, NPIX * 32);
  k_zero_bf16<<<2048, 256, 0, stream>>>(H2, NPIX * 32);
  k_zero_bf16<<<2048, 256, 0, stream>>>(H3, NPIX * 16);

  // pack conv + MLP weights into B-fragment layout
  auto pack = [&](const float* W, bf16* dst, int Cout, int Cin, int KH, int KW, int CP) {
    long total = (long)((Cout + 15) / 16) * KH * KW * (CP / 32) * 512;
    int blocks = (int)((total + 255) / 256);
    k_pack_B<<<blocks, 256, 0, stream>>>(W, dst, Cout, Cin, KH, KW, CP);
  };
  pack(F(43), pg0, 64, 35, 3, 3, 64);
  pack(F(45), pc0, 32, 35, 3, 3, 64);
  pack(F(47), pg1, 64, 64, 5, 5, 64);
  pack(F(49), pc1, 32, 64, 5, 5, 64);
  pack(F(51), pg2, 32, 48, 3, 3, 64);
  pack(F(53), pc2, 16, 48, 3, 3, 64);
  pack(F(69), pW1, 256, 224, 1, 1, 224);

  // trajectory GRU stacks (finals: na -> all 3 layers, ta/aa -> last layer)
  k_gru<<<1, 384, 0, stream>>>(x,    0, 3,  F(7),  F(8),  F(9),  F(10), seqA, naF);
  k_gru<<<1, 384, 0, stream>>>(seqA, 1, 16, F(11), F(12), F(13), F(14), seqB, naF + 368);
  k_gru<<<1, 384, 0, stream>>>(seqB, 1, 16, F(15), F(16), F(17), F(18), nullptr, naF + 736);
  k_gru<<<1, 384, 0, stream>>>(x,    0, 3,  F(19), F(20), F(21), F(22), seqA, taF);
  k_gru<<<1, 384, 0, stream>>>(seqA, 1, 16, F(23), F(24), F(25), F(26), seqB, taF);
  k_gru<<<1, 384, 0, stream>>>(seqB, 1, 16, F(27), F(28), F(29), F(30), nullptr, taF);
  k_gru<<<1, 384, 0, stream>>>(x,    2, 6,  F(31), F(32), F(33), F(34), seqA, aaF);
  k_gru<<<1, 384, 0, stream>>>(seqA, 1, 16, F(35), F(36), F(37), F(38), seqB, aaF);
  k_gru<<<1, 384, 0, stream>>>(seqB, 1, 16, F(39), F(40), F(41), F(42), nullptr, aaF);

  // ConvGRU video branch: 8 frames x 3 layers of (build, gates, cand)
  const int NB = 576;  // 147456 pixels / 256
  for (int t = 0; t < 8; ++t) {
    // layer 0: Cin_x=3, hd=32, k=3
    k_build_l0<<<NB, 256, 0, stream>>>(img, t, H1, INA, INB);
    k_conv_wmma<3, 3, 0><<<dim3(576, 4, 4), 128, 9 * 2 * 1024, stream>>>(
        INA, pg0, F(44), 32, 3, Zb, INB, H1);
    k_conv_wmma<3, 3, 1><<<dim3(576, 2, 4), 128, 9 * 2 * 1024, stream>>>(
        INB, pc0, F(46), 32, 3, Zb, nullptr, H1);
    // layer 1: Cin_x=32, hd=32, k=5
    k_build_hh<<<NB, 256, 0, stream>>>(H1, 32, H2, 32, INA, INB);
    k_conv_wmma<5, 5, 0><<<dim3(576, 4, 4), 128, 25 * 2 * 1024, stream>>>(
        INA, pg1, F(48), 32, 32, Zb, INB, H2);
    k_conv_wmma<5, 5, 1><<<dim3(576, 2, 4), 128, 25 * 2 * 1024, stream>>>(
        INB, pc1, F(50), 32, 32, Zb, nullptr, H2);
    // layer 2: Cin_x=32, hd=16, k=3
    k_build_hh<<<NB, 256, 0, stream>>>(H2, 32, H3, 16, INA, INB);
    k_conv_wmma<3, 3, 0><<<dim3(576, 2, 4), 128, 9 * 2 * 1024, stream>>>(
        INA, pg2, F(52), 16, 32, Zb, INB, H3);
    k_conv_wmma<3, 3, 1><<<dim3(576, 1, 4), 128, 9 * 2 * 1024, stream>>>(
        INB, pc2, F(54), 16, 32, Zb, nullptr, H3);
  }

  // pooling + graph + prediction head
  k_pool<<<36, 256, 0, stream>>>(H3, vfeat);
  k_node_in<<<(23 * 192 + 255) / 256, 256, 0, stream>>>(naF, vfeat, imginds, node_in);
  k_sage1<<<23, 256, 0, stream>>>(node_in, aaF,
                                  F(55), F(56), F(57), F(58), F(59), F(60),
                                  F(61), F(62), F(63), F(64), F(65), F(66), h1);
  k_sage2<<<23, 256, 0, stream>>>(h1, F(67), F(68), gf);
  k_union<<<(int)(((long)NQ * 224 + 255) / 256), 256, 0, stream>>>(gf, taF, q_from, q_to, U);
  k_mlp1<<<dim3(64, 16), 128, 0, stream>>>(U, pW1, F(70), A1);
  k_mlp2<<<16, 256, 0, stream>>>(A1, F(71), F(72), (float*)d_out);
}
